// MHCA_2473901163150
// MI455X (gfx1250) — matrix operand order repaired and checked by reading, not compiled, not run on hardware
//
#include <hip/hip_runtime.h>
#include <hip/hip_bf16.h>
#include <math.h>

typedef __attribute__((ext_vector_type(2)))  float  v2f;
typedef __attribute__((ext_vector_type(8)))  float  v8f;
typedef __attribute__((ext_vector_type(8)))  __bf16 v8bf;
typedef __attribute__((ext_vector_type(16))) __bf16 v16bf;
typedef __attribute__((ext_vector_type(4)))  unsigned int u32x4;
typedef __attribute__((ext_vector_type(8)))  int i32x8;
typedef __attribute__((ext_vector_type(4)))  int i32x4;

#define NT 3136          // h*w = 56*56
#define HWW 56
#define CDIM 256
#define C3 768
#define HEADS 8
#define HD 32
#define BATCH 16
#define LDSPITCH 40      // bf16 row pitch (80B: 16B-aligned, conflict-reducing)

static __device__ __forceinline__ v8f v8zero() {
    v8f z = {0.f,0.f,0.f,0.f,0.f,0.f,0.f,0.f};
    return z;
}

static __device__ __forceinline__ void bf16split(float x, __bf16& h, __bf16& l) {
    h = (__bf16)x;
    l = (__bf16)(x - (float)h);
}

static __device__ __forceinline__ v16bf ldsfrag(const __bf16* p, int off0, int off1) {
    v8bf a = *(const v8bf*)(p + off0);
    v8bf b = *(const v8bf*)(p + off1);
    return __builtin_shufflevector(a, b, 0,1,2,3,4,5,6,7,8,9,10,11,12,13,14,15);
}

// ---------------------------------------------------------------------------
// bf16-split WMMA GEMM: Y[b][m][n] = sum_k A[m][k] * X[b][k][n]  (f32 in/out)
// x = hi + lo (bf16 each); D += Ahi*Bhi + Ahi*Blo + Alo*Bhi  (f32 accum)
// Tile 128(M) x 64(N) per 256-thread WG; K staged 32 wide (= one wmma K).
// ---------------------------------------------------------------------------
__global__ __launch_bounds__(256)
void wmma_gemm_kernel(const float* __restrict__ A,   // [M][K], K contiguous
                      const float* __restrict__ X,   // [B][K][NT]
                      float* __restrict__ Y,         // [B][M][NT]
                      int M, int K)
{
    const int nBase = blockIdx.x * 64;
    const int mBase = blockIdx.y * 128;
    const int batch = blockIdx.z;
    const float* Xb = X + (size_t)batch * K * NT;
    float* Yb = Y + (size_t)batch * M * NT;

    __shared__ __align__(16) __bf16 lAhi[128 * LDSPITCH]; // [m][k]
    __shared__ __align__(16) __bf16 lAlo[128 * LDSPITCH];
    __shared__ __align__(16) __bf16 lBhi[64 * LDSPITCH];  // [n][k] (transposed)
    __shared__ __align__(16) __bf16 lBlo[64 * LDSPITCH];

    const int t     = threadIdx.x;
    const int lane  = t & 31;
    const int wave  = t >> 5;
    const int mRow  = lane & 15;          // A row / B,D col within 16-tile
    const int g0    = (lane >> 4) * 8;    // A-frag K octet select (ISA layout)
    const int h0    = (lane >> 4) * 8;    // B-frag K octet select (ISA layout)
    const int dRow  = 8 * (lane >> 4);    // D-row offset

    v8f acc[4];
    #pragma unroll
    for (int i = 0; i < 4; ++i) acc[i] = v8zero();

    for (int k0 = 0; k0 < K; k0 += 32) {
        // Stage A tile 128x32: global coalesced along k, split into hi/lo
        #pragma unroll
        for (int i = 0; i < 16; ++i) {
            int idx = t + i * 256;
            int kIn = idx & 31;
            int mIn = idx >> 5;
            float xv = A[(size_t)(mBase + mIn) * K + (k0 + kIn)];
            __bf16 h, l; bf16split(xv, h, l);
            lAhi[mIn * LDSPITCH + kIn] = h;
            lAlo[mIn * LDSPITCH + kIn] = l;
        }
        // Stage B tile 32x64: global coalesced along n, transpose to [n][k]
        #pragma unroll
        for (int i = 0; i < 8; ++i) {
            int idx = t + i * 256;
            int nIn = idx & 63;
            int kIn = idx >> 6;
            float xv = Xb[(size_t)(k0 + kIn) * NT + (nBase + nIn)];
            __bf16 h, l; bf16split(xv, h, l);
            lBhi[nIn * LDSPITCH + kIn] = h;
            lBlo[nIn * LDSPITCH + kIn] = l;
        }
        __syncthreads();

        const __bf16* aHiRow = lAhi + (wave * 16 + mRow) * LDSPITCH;
        const __bf16* aLoRow = lAlo + (wave * 16 + mRow) * LDSPITCH;
        v16bf ahi = ldsfrag(aHiRow, g0, 16 + g0);
        v16bf alo = ldsfrag(aLoRow, g0, 16 + g0);
        #pragma unroll
        for (int nt = 0; nt < 4; ++nt) {
            const __bf16* bHiRow = lBhi + (nt * 16 + mRow) * LDSPITCH + h0;
            const __bf16* bLoRow = lBlo + (nt * 16 + mRow) * LDSPITCH + h0;
            v16bf bhi = ldsfrag(bHiRow, 0, 16);
            v16bf blo = ldsfrag(bLoRow, 0, 16);
            acc[nt] = __builtin_amdgcn_wmma_f32_16x16x32_bf16(
                false, ahi, false, bhi, (short)0, acc[nt], false, false);
            acc[nt] = __builtin_amdgcn_wmma_f32_16x16x32_bf16(
                false, ahi, false, blo, (short)0, acc[nt], false, false);
            acc[nt] = __builtin_amdgcn_wmma_f32_16x16x32_bf16(
                false, alo, false, bhi, (short)0, acc[nt], false, false);
        }
        __syncthreads();
    }

    #pragma unroll
    for (int nt = 0; nt < 4; ++nt) {
        int n = nBase + nt * 16 + mRow;
        #pragma unroll
        for (int i = 0; i < 8; ++i) {
            int m = mBase + wave * 16 + i + dRow;
            Yb[(size_t)m * NT + n] = acc[nt][i];
        }
    }
}

// ---------------------------------------------------------------------------
// Depthwise 3x3 conv, pad=1. grid: (ceil(3136/256), 768, 16)
// ---------------------------------------------------------------------------
__global__ __launch_bounds__(256)
void dwconv3x3_kernel(const float* __restrict__ in,    // [B][768][NT]
                      const float* __restrict__ wdw,   // [768][9]
                      float* __restrict__ out)         // [B][768][NT]
{
    const int hw = blockIdx.x * blockDim.x + threadIdx.x;
    if (hw >= NT) return;
    const int ch = blockIdx.y;
    const int b  = blockIdx.z;

    const float* w = wdw + ch * 9;   // uniform per block -> scalar loads
    const float* p = in + ((size_t)b * C3 + ch) * NT;

    const int y = hw / HWW;
    const int x = hw - y * HWW;

    float s = 0.f;
    #pragma unroll
    for (int dy = -1; dy <= 1; ++dy) {
        #pragma unroll
        for (int dx = -1; dx <= 1; ++dx) {
            int yy = y + dy, xx = x + dx;
            if (yy >= 0 && yy < HWW && xx >= 0 && xx < HWW)
                s += w[(dy + 1) * 3 + (dx + 1)] * p[yy * HWW + xx];
        }
    }
    out[((size_t)b * C3 + ch) * NT + hw] = s;
}

// ---------------------------------------------------------------------------
// TDM: load one 3D tile [2 planes (q,k)] x [32 rows] x [16 f32] into LDS.
// plane stride = 256*NT floats, row stride = NT floats. Per-wave, uniform.
// D# packing per CDNA5 ISA 8.3-8.5 (count=1, data_size=4B, type=2).
// 6-arg builtin form (this toolchain): extra i32x8 beyond architectural D#
// groups (VADDR4 path, unused) is zero-filled.
// ---------------------------------------------------------------------------
static __device__ __forceinline__ void tdm_load_qk(const float* gaddr, unsigned ldsOff) {
    unsigned long long ga = (unsigned long long)(uintptr_t)gaddr;
    unsigned galo = __builtin_amdgcn_readfirstlane((unsigned)(ga & 0xffffffffu));
    unsigned gahi = __builtin_amdgcn_readfirstlane((unsigned)(ga >> 32));
    unsigned lds  = __builtin_amdgcn_readfirstlane(ldsOff);

    u32x4 g0;
    g0[0] = 1u;                                   // count=1, user desc
    g0[1] = lds;                                  // lds_addr (bytes)
    g0[2] = galo;                                 // global_addr[31:0]
    g0[3] = (gahi & 0x1ffffffu) | (2u << 30);     // global_addr[56:32] | type=2

    const unsigned s1 = 256u * NT;                // plane stride (dim1_stride)
    i32x8 g1;
    g1[0] = (2 << 16);                            // data_size = 4B
    g1[1] = (int)((NT & 0xffff) << 16);           // tensor_dim0[15:0]
    g1[2] = (NT >> 16) | (32 << 16);              // tensor_dim0[31:16] | tensor_dim1[15:0]
    g1[3] = (16 << 16);                           // tensor_dim1[31:16]=0 | tile_dim0=16
    g1[4] = 32 | (2 << 16);                       // tile_dim1=32 | tile_dim2=2
    g1[5] = NT;                                   // tensor_dim0_stride[31:0]
    g1[6] = (int)((s1 & 0xffffu) << 16);          // stride0[47:32]=0 | stride1[15:0]
    g1[7] = (int)(s1 >> 16);                      // stride1[47:16]
    i32x4 g2 = {2, 0, 0, 0};                      // tensor_dim2 = 2
    i32x4 g3 = {0, 0, 0, 0};
    i32x8 g4 = {0, 0, 0, 0, 0, 0, 0, 0};          // unused trailing group
    __builtin_amdgcn_tensor_load_to_lds(g0, g1, g2, g3, g4, 0);
}

// ---------------------------------------------------------------------------
// Channel attention per (b, head): exact f32 WMMA + TDM-staged q/k chunks.
// ---------------------------------------------------------------------------
__global__ __launch_bounds__(256)
void attn_kernel(const float* __restrict__ qkv,   // [B][768][NT] (q|k|v)
                 const float* __restrict__ temp,  // [8]
                 float* __restrict__ out)         // [B][256][NT]
{
    const int b    = blockIdx.x >> 3;
    const int head = blockIdx.x & 7;
    const float* q = qkv + ((size_t)b * C3 + head * HD) * NT;
    const float* k = q + (size_t)CDIM * NT;
    const float* v = q + (size_t)2 * CDIM * NT;
    float* o = out + ((size_t)b * CDIM + head * HD) * NT;

    __shared__ float sS[32][32];
    __shared__ float sNorm[64];
    __shared__ float sA[32][33];
    __shared__ __align__(16) float stage[8][2][2 * 32 * 16]; // [wave][buf][qk|row|16]

    const int t     = threadIdx.x;
    const int lane  = t & 31;
    const int wave  = t >> 5;
    const int mRow  = lane & 15;
    const int kHalf = (lane >> 4) * 2;
    const int dRow  = 8 * (lane >> 4);

    for (int i = t; i < 32 * 32; i += 256) ((float*)sS)[i] = 0.f;

    // -------- Phase 1: inverse L2 norms (64 rows = 32 q + 32 k) -----------
    #pragma unroll
    for (int rr = 0; rr < 8; ++rr) {
        const int r = wave * 8 + rr;
        const float* src = (r < 32) ? (q + (size_t)r * NT)
                                    : (k + (size_t)(r - 32) * NT);
        float s = 0.f;
        for (int j = lane; j < NT; j += 32) { float x = src[j]; s += x * x; }
        #pragma unroll
        for (int off = 16; off > 0; off >>= 1) s += __shfl_xor(s, off, 32);
        if (lane == 0) sNorm[r] = 1.0f / fmaxf(sqrtf(s), 1e-12f);
    }
    __syncthreads();

    // -------- Phase 2: S = q @ k^T, K split across waves, TDM staged ------
    const unsigned ldsB0 = (unsigned)(uintptr_t)&stage[wave][0][0];
    const unsigned ldsB1 = (unsigned)(uintptr_t)&stage[wave][1][0];
    v8f accS[4];
    #pragma unroll
    for (int i = 0; i < 4; ++i) accS[i] = v8zero();

    tdm_load_qk(q + (size_t)wave * 16, ldsB0);        // prologue: first chunk
    int buf = 0;
    for (int cc = wave; cc < NT / 16; cc += 8) {
        if (cc + 8 < NT / 16) {                        // prefetch next chunk
            tdm_load_qk(q + (size_t)(cc + 8) * 16, buf ? ldsB0 : ldsB1);
            __builtin_amdgcn_s_wait_tensorcnt((short)1);
        } else {
            __builtin_amdgcn_s_wait_tensorcnt((short)0);
        }
        const float* sq = &stage[wave][buf][0];
        const float* sk = sq + 32 * 16;
        #pragma unroll
        for (int kk = 0; kk < 16; kk += 4) {
            v2f aq[2], bk[2];
            #pragma unroll
            for (int mt = 0; mt < 2; ++mt) {
                aq[mt].x = sq[(mt * 16 + mRow) * 16 + kk + kHalf];
                aq[mt].y = sq[(mt * 16 + mRow) * 16 + kk + kHalf + 1];
                bk[mt].x = sk[(mt * 16 + mRow) * 16 + kk + kHalf];
                bk[mt].y = sk[(mt * 16 + mRow) * 16 + kk + kHalf + 1];
            }
            #pragma unroll
            for (int mt = 0; mt < 2; ++mt)
                #pragma unroll
                for (int nt = 0; nt < 2; ++nt)
                    accS[mt * 2 + nt] = __builtin_amdgcn_wmma_f32_16x16x4_f32(
                        false, aq[mt], false, bk[nt], (short)0,
                        accS[mt * 2 + nt], false, false);
        }
        buf ^= 1;
    }
    #pragma unroll
    for (int tile = 0; tile < 4; ++tile) {
        const int mt = tile >> 1, nt = tile & 1;
        #pragma unroll
        for (int i = 0; i < 8; ++i)
            atomicAdd(&sS[mt * 16 + i + dRow][nt * 16 + mRow], accS[tile][i]);
    }
    __syncthreads();

    // -------- Phase 3: scaled softmax rows --------------------------------
    if (t < 32) {
        const float tscale = temp[head];
        const float iq = sNorm[t];
        float row[32];
        float mx = -1e30f;
        #pragma unroll
        for (int d = 0; d < 32; ++d) {
            float lg = sS[t][d] * iq * sNorm[32 + d] * tscale;
            row[d] = lg; mx = fmaxf(mx, lg);
        }
        float ssum = 0.f;
        #pragma unroll
        for (int d = 0; d < 32; ++d) { float e = __expf(row[d] - mx); row[d] = e; ssum += e; }
        const float inv = 1.f / ssum;
        #pragma unroll
        for (int d = 0; d < 32; ++d) sA[t][d] = row[d] * inv;
    }
    __syncthreads();

    // -------- Phase 4: out = A(32x32) @ v(32xNT), B-frag reused -----------
    for (int ntv = wave; ntv < NT / 16; ntv += 8) {
        const int n = ntv * 16 + mRow;
        v8f a0 = v8zero(), a1 = v8zero();
        #pragma unroll
        for (int ks = 0; ks < 32; ks += 4) {
            v2f bf;
            bf.x = v[(size_t)(ks + kHalf) * NT + n];
            bf.y = v[(size_t)(ks + kHalf + 1) * NT + n];
            v2f f0, f1;
            f0.x = sA[mRow][ks + kHalf];      f0.y = sA[mRow][ks + kHalf + 1];
            f1.x = sA[16 + mRow][ks + kHalf]; f1.y = sA[16 + mRow][ks + kHalf + 1];
            a0 = __builtin_amdgcn_wmma_f32_16x16x4_f32(false, f0, false, bf, (short)0, a0, false, false);
            a1 = __builtin_amdgcn_wmma_f32_16x16x4_f32(false, f1, false, bf, (short)0, a1, false, false);
        }
        #pragma unroll
        for (int i = 0; i < 8; ++i) {
            o[(size_t)(i + dRow) * NT + n]      = a0[i];
            o[(size_t)(16 + i + dRow) * NT + n] = a1[i];
        }
    }
}

// ---------------------------------------------------------------------------
extern "C" void kernel_launch(void* const* d_in, const int* in_sizes, int n_in,
                              void* d_out, int out_size, void* d_ws, size_t ws_size,
                              hipStream_t stream) {
    const float* x           = (const float*)d_in[0];
    const float* w_qkv       = (const float*)d_in[1];
    const float* w_dw        = (const float*)d_in[2];
    const float* w_proj      = (const float*)d_in[3];
    const float* temperature = (const float*)d_in[4];
    float* out = (float*)d_out;

    const size_t qkvBytes = (size_t)BATCH * C3 * NT * sizeof(float);
    float* qkv_raw = (float*)d_ws;
    float* qkv_dw  = (float*)((char*)d_ws + qkvBytes);
    float* ctx     = qkv_raw;   // raw qkv dead after depthwise -> reuse

    dim3 blk(256);
    wmma_gemm_kernel<<<dim3(NT / 64, C3 / 128, BATCH), blk, 0, stream>>>(
        w_qkv, x, qkv_raw, C3, CDIM);
    dwconv3x3_kernel<<<dim3((NT + 255) / 256, C3, BATCH), blk, 0, stream>>>(
        qkv_raw, w_dw, qkv_dw);
    attn_kernel<<<dim3(BATCH * HEADS), blk, 0, stream>>>(qkv_dw, temperature, ctx);
    wmma_gemm_kernel<<<dim3(NT / 64, CDIM / 128, BATCH), blk, 0, stream>>>(
        w_proj, ctx, out, CDIM, CDIM);
}
